// SelfAttention_56057913147530
// MI455X (gfx1250) — matrix operand order repaired and checked by reading, not compiled, not run on hardware
//
#include <hip/hip_runtime.h>

#define DEV __device__ __forceinline__

typedef __attribute__((ext_vector_type(16))) __bf16 v16bf;
typedef __attribute__((ext_vector_type(8)))  float  v8f;

constexpr int Bb = 4, Ss = 4096, Dd = 1024;
constexpr int BM = 128, BN = 128, BK = 32;
constexpr int LDT = BK + 4;              // padded LDS row (36 bf16 = 72B) to spread banks
constexpr int TILE = BM * LDT;           // one LDS tile (elements)
constexpr float SCALE = 0.03125f;        // 1/sqrt(1024)

union FragU { v16bf v; unsigned u[8]; };
union PkU   { __bf16 h[2]; unsigned u; };

DEV unsigned pack_bf16(float a, float b) {
  PkU p; p.h[0] = (__bf16)a; p.h[1] = (__bf16)b; return p.u;
}

// A fragment: 16x32 bf16, per ISA layout: lanes 0-15 row m=lane, K={0..7,16..23};
// lanes 16-31 row m=lane-16, K={8..15,24..31}; pairs packed per dword.
DEV v16bf frag_a(const __bf16* lds, int rowBase, int lane) {
  int half = lane >> 4, mr = lane & 15;
  const __bf16* rp = lds + (rowBase + mr) * LDT;
  FragU f;
#pragma unroll
  for (int v = 0; v < 8; ++v) {
    int k = (v < 4 ? 2 * v : 8 + 2 * v) + 8 * half;
    f.u[v] = *(const unsigned*)(rp + k);
  }
  return f.v;
}

// B fragment: 32x16 bf16: lane n=lane&15; lanes 0-15 hold K={0..7,16..23}, lanes 16-31 K={8..15,24..31}.
DEV v16bf frag_b(const __bf16* lds, int colBase, int lane) {
  int half = lane >> 4, n = lane & 15;
  const __bf16* rp = lds + (colBase + n) * LDT + 8 * half;
  FragU f;
#pragma unroll
  for (int v = 0; v < 8; ++v) f.u[v] = *(const unsigned*)(rp + (v < 4 ? 2 * v : 8 + 2 * v));
  return f.v;
}

// ---- register-staged tiles (256 threads cover a 128x32 tile; 16 elems/thread) ----

struct RawB { uint4 a, b; };             // 16 bf16 in regs
struct RawF { float4 f0, f1, f2, f3; };  // 16 fp32 in regs

DEV RawB load_g_bf16(const __bf16* __restrict__ src, int ldsrc, int tid) {
  int row = tid >> 1, cb = (tid & 1) * 16;
  const uint4* p = (const uint4*)(src + (size_t)row * ldsrc + cb);
  RawB t; t.a = p[0]; t.b = p[1]; return t;
}

DEV RawF load_g_f32(const float* __restrict__ src, int ldsrc, int tid) {
  int row = tid >> 1, cb = (tid & 1) * 16;
  const float* p = src + (size_t)row * ldsrc + cb;
  RawF t;
  t.f0 = *(const float4*)(p);
  t.f1 = *(const float4*)(p + 4);
  t.f2 = *(const float4*)(p + 8);
  t.f3 = *(const float4*)(p + 12);
  return t;
}

DEV void lds_store_bf16(__bf16* lds, RawB t, int tid) {
  int row = tid >> 1, cb = (tid & 1) * 16;
  unsigned* d = (unsigned*)(lds + row * LDT + cb);
  d[0] = t.a.x; d[1] = t.a.y; d[2] = t.a.z; d[3] = t.a.w;
  d[4] = t.b.x; d[5] = t.b.y; d[6] = t.b.z; d[7] = t.b.w;
}

DEV void lds_store_f32(__bf16* lds, RawF t, int tid) {
  int row = tid >> 1, cb = (tid & 1) * 16;
  unsigned* d = (unsigned*)(lds + row * LDT + cb);
  d[0] = pack_bf16(t.f0.x, t.f0.y); d[1] = pack_bf16(t.f0.z, t.f0.w);
  d[2] = pack_bf16(t.f1.x, t.f1.y); d[3] = pack_bf16(t.f1.z, t.f1.w);
  d[4] = pack_bf16(t.f2.x, t.f2.y); d[5] = pack_bf16(t.f2.z, t.f2.w);
  d[6] = pack_bf16(t.f3.x, t.f3.y); d[7] = pack_bf16(t.f3.z, t.f3.w);
}

// fused softmax numerator: lds = exp(val - m_row)
DEV void lds_store_exp(__bf16* lds, RawB t, float mrow, int tid) {
  int row = tid >> 1, cb = (tid & 1) * 16;
  unsigned raw[8] = {t.a.x, t.a.y, t.a.z, t.a.w, t.b.x, t.b.y, t.b.z, t.b.w};
  unsigned* d = (unsigned*)(lds + row * LDT + cb);
#pragma unroll
  for (int i = 0; i < 8; ++i) {
    PkU in; in.u = raw[i];
    d[i] = pack_bf16(__expf((float)in.h[0] - mrow), __expf((float)in.h[1] - mrow));
  }
}

// one K-step: each wave does 2x4 = 8 WMMA tiles (32x64 region)
DEV void mma_step(const __bf16* ldsA, const __bf16* ldsB, int waveM, int waveN,
                  int lane, v8f acc[2][4]) {
  v16bf av[2], bv[4];
#pragma unroll
  for (int i = 0; i < 2; ++i) av[i] = frag_a(ldsA, waveM * 32 + i * 16, lane);
#pragma unroll
  for (int j = 0; j < 4; ++j) bv[j] = frag_b(ldsB, waveN * 64 + j * 16, lane);
#pragma unroll
  for (int i = 0; i < 2; ++i)
#pragma unroll
    for (int j = 0; j < 4; ++j)
      acc[i][j] = __builtin_amdgcn_wmma_f32_16x16x32_bf16(
          false, av[i], false, bv[j], (short)0, acc[i][j], false, false);
}

// ---- kernel 1: Q/K/V projections (x fp32 -> bf16 Q,K and transposed V) ----
__global__ __launch_bounds__(256) void qkv_kernel(
    const float* __restrict__ x, const float* __restrict__ Wq,
    const float* __restrict__ Wk, const float* __restrict__ Wv,
    __bf16* __restrict__ Qh, __bf16* __restrict__ Kh, __bf16* __restrict__ Vt) {
  __shared__ __bf16 ldsA[2 * TILE];
  __shared__ __bf16 ldsB[2 * TILE];
  int tid = threadIdx.x, lane = tid & 31, wave = tid >> 5;
  int waveM = wave >> 1, waveN = wave & 1;
  int m0 = blockIdx.x * BM;            // flat (b*S+s) row
  int n0 = blockIdx.y * BN;            // output feature
  int sel = blockIdx.z;                // 0:Q 1:K 2:V
  const float* W = (sel == 0) ? Wq : (sel == 1 ? Wk : Wv);
  const float* As = x + (size_t)m0 * Dd;
  const float* Bs = W + (size_t)n0 * Dd;
  v8f acc[2][4] = {};
  RawF ra = load_g_f32(As, Dd, tid);
  RawF rb = load_g_f32(Bs, Dd, tid);
  int buf = 0;
  for (int k0 = 0; k0 < Dd; k0 += BK) {
    lds_store_f32(ldsA + buf * TILE, ra, tid);
    lds_store_f32(ldsB + buf * TILE, rb, tid);
    __syncthreads();
    if (k0 + BK < Dd) {                // in-flight loads for next tile overlap WMMA
      ra = load_g_f32(As + k0 + BK, Dd, tid);
      rb = load_g_f32(Bs + k0 + BK, Dd, tid);
    }
    mma_step(ldsA + buf * TILE, ldsB + buf * TILE, waveM, waveN, lane, acc);
    buf ^= 1;
  }
  int half = lane >> 4, nL = lane & 15;
  int mBase = m0 + waveM * 32, nBase = n0 + waveN * 64;
  if (sel == 2) {                      // V stored transposed per batch: Vt[b][e][s]
#pragma unroll
    for (int i = 0; i < 2; ++i)
#pragma unroll
      for (int j = 0; j < 4; ++j) {
        int gn = nBase + j * 16 + nL;
#pragma unroll
        for (int r = 0; r < 8; ++r) {
          int gm = mBase + i * 16 + 8 * half + r;
          int bb = gm >> 12, ss = gm & (Ss - 1);
          Vt[(size_t)bb * Dd * Ss + (size_t)gn * Ss + ss] = (__bf16)acc[i][j][r];
        }
      }
  } else {
    __bf16* O = (sel == 0) ? Qh : Kh;
#pragma unroll
    for (int i = 0; i < 2; ++i)
#pragma unroll
      for (int j = 0; j < 4; ++j) {
        int gn = nBase + j * 16 + nL;
#pragma unroll
        for (int r = 0; r < 8; ++r) {
          int gm = mBase + i * 16 + 8 * half + r;
          O[(size_t)gm * Dd + gn] = (__bf16)acc[i][j][r];
        }
      }
  }
}

// ---- kernel 2: SC = scale * Q K^T with causal mask, bf16 ----
__global__ __launch_bounds__(256) void scores_kernel(
    const __bf16* __restrict__ Qh, const __bf16* __restrict__ Kh,
    __bf16* __restrict__ SC) {
  int tid = threadIdx.x, lane = tid & 31, wave = tid >> 5;
  int q0 = blockIdx.x * BM, k0n = blockIdx.y * BN, b = blockIdx.z;
  __bf16* sc = SC + (size_t)b * Ss * Ss;
  if (k0n > q0 + (BM - 1)) {           // fully above diagonal: just write -inf
    int row = tid >> 1, cbase = (tid & 1) * 64;
    unsigned* d = (unsigned*)(sc + (size_t)(q0 + row) * Ss + k0n + cbase);
#pragma unroll
    for (int i = 0; i < 32; ++i) d[i] = 0xFF80FF80u;  // two bf16 -inf
    return;
  }
  __shared__ __bf16 ldsA[2 * TILE];
  __shared__ __bf16 ldsB[2 * TILE];
  const __bf16* As = Qh + (size_t)b * Ss * Dd + (size_t)q0  * Dd;
  const __bf16* Bs = Kh + (size_t)b * Ss * Dd + (size_t)k0n * Dd;
  int waveM = wave >> 1, waveN = wave & 1;
  v8f acc[2][4] = {};
  RawB ra = load_g_bf16(As, Dd, tid);
  RawB rb = load_g_bf16(Bs, Dd, tid);
  int buf = 0;
  for (int k0 = 0; k0 < Dd; k0 += BK) {
    lds_store_bf16(ldsA + buf * TILE, ra, tid);
    lds_store_bf16(ldsB + buf * TILE, rb, tid);
    __syncthreads();
    if (k0 + BK < Dd) {
      ra = load_g_bf16(As + k0 + BK, Dd, tid);
      rb = load_g_bf16(Bs + k0 + BK, Dd, tid);
    }
    mma_step(ldsA + buf * TILE, ldsB + buf * TILE, waveM, waveN, lane, acc);
    buf ^= 1;
  }
  bool needMask = (k0n + (BN - 1) > q0);
  int half = lane >> 4, nL = lane & 15;
#pragma unroll
  for (int i = 0; i < 2; ++i)
#pragma unroll
    for (int j = 0; j < 4; ++j) {
      int gk = k0n + waveN * 64 + j * 16 + nL;
#pragma unroll
      for (int r = 0; r < 8; ++r) {
        int gq = q0 + waveM * 32 + i * 16 + 8 * half + r;
        float vv = acc[i][j][r] * SCALE;
        vv = (needMask && gk > gq) ? -__builtin_inff() : vv;  // branchless select
        sc[(size_t)gq * Ss + gk] = (__bf16)vv;
      }
    }
}

// ---- kernel 3: per-row softmax stats (m, 1/sum(exp)) ----
__global__ __launch_bounds__(256) void stats_kernel(
    const __bf16* __restrict__ SC, float* __restrict__ stats) {
  int tid = threadIdx.x;
  size_t row = blockIdx.x;             // flat (b*S + q)
  const uint4* p = (const uint4*)(SC + row * Ss);
  uint4 a = p[tid * 2], b = p[tid * 2 + 1];
  unsigned t[8] = {a.x, a.y, a.z, a.w, b.x, b.y, b.z, b.w};
  float vals[16];
#pragma unroll
  for (int i = 0; i < 8; ++i) {
    PkU u; u.u = t[i];
    vals[2 * i] = (float)u.h[0]; vals[2 * i + 1] = (float)u.h[1];
  }
  float pm = vals[0];
#pragma unroll
  for (int i = 1; i < 16; ++i) pm = fmaxf(pm, vals[i]);
  __shared__ float red[256];
  red[tid] = pm; __syncthreads();
  for (int s2 = 128; s2 > 0; s2 >>= 1) {
    if (tid < s2) red[tid] = fmaxf(red[tid], red[tid + s2]);
    __syncthreads();
  }
  float m = red[0]; __syncthreads();
  float ps = 0.f;
#pragma unroll
  for (int i = 0; i < 16; ++i) ps += __expf(vals[i] - m);
  red[tid] = ps; __syncthreads();
  for (int s2 = 128; s2 > 0; s2 >>= 1) {
    if (tid < s2) red[tid] += red[tid + s2];
    __syncthreads();
  }
  if (tid == 0) { stats[2 * row] = m; stats[2 * row + 1] = 1.0f / red[0]; }
}

// ---- kernel 4: O = diag(1/l) * exp(SC - m) * V^T (exp fused into staging) ----
__global__ __launch_bounds__(256) void pv_kernel(
    const __bf16* __restrict__ SC, const float* __restrict__ stats,
    const __bf16* __restrict__ Vt, float* __restrict__ out) {
  int tid = threadIdx.x, lane = tid & 31, wave = tid >> 5;
  int q0 = blockIdx.x * BM, e0 = blockIdx.y * BN, b = blockIdx.z;
  const __bf16* As = SC + (size_t)b * Ss * Ss + (size_t)q0 * Ss;
  const __bf16* Bs = Vt + (size_t)b * Dd * Ss + (size_t)e0 * Ss;
  __shared__ __bf16 ldsA[2 * TILE];
  __shared__ __bf16 ldsB[2 * TILE];
  int waveM = wave >> 1, waveN = wave & 1;
  float mrow = stats[2 * ((size_t)b * Ss + q0 + (tid >> 1))];
  v8f acc[2][4] = {};
  int kend = (q0 + BM < Ss) ? (q0 + BM) : Ss;   // causal clip: P==0 beyond q0+BM
  RawB ra = load_g_bf16(As, Ss, tid);
  RawB rb = load_g_bf16(Bs, Ss, tid);
  int buf = 0;
  for (int k0 = 0; k0 < kend; k0 += BK) {
    lds_store_exp (ldsA + buf * TILE, ra, mrow, tid);
    lds_store_bf16(ldsB + buf * TILE, rb, tid);
    __syncthreads();
    if (k0 + BK < kend) {
      ra = load_g_bf16(As + k0 + BK, Ss, tid);
      rb = load_g_bf16(Bs + k0 + BK, Ss, tid);
    }
    mma_step(ldsA + buf * TILE, ldsB + buf * TILE, waveM, waveN, lane, acc);
    buf ^= 1;
  }
  int half = lane >> 4, nL = lane & 15;
#pragma unroll
  for (int i = 0; i < 2; ++i)
#pragma unroll
    for (int r = 0; r < 8; ++r) {
      int gq = q0 + waveM * 32 + i * 16 + 8 * half + r;
      float il = stats[2 * ((size_t)b * Ss + gq) + 1];
#pragma unroll
      for (int j = 0; j < 4; ++j) {
        int ge = e0 + waveN * 64 + j * 16 + nL;
        out[(size_t)b * Ss * Dd + (size_t)gq * Dd + ge] = acc[i][j][r] * il;
      }
    }
}

extern "C" void kernel_launch(void* const* d_in, const int* in_sizes, int n_in,
                              void* d_out, int out_size, void* d_ws, size_t ws_size,
                              hipStream_t stream) {
  const float* x  = (const float*)d_in[0];
  const float* Wq = (const float*)d_in[1];
  const float* Wk = (const float*)d_in[2];
  const float* Wv = (const float*)d_in[3];
  char* w = (char*)d_ws;
  __bf16* Qh = (__bf16*)(w);                                   // 32 MB
  __bf16* Kh = (__bf16*)(w + (size_t)32 * 1024 * 1024);        // 32 MB
  __bf16* Vt = (__bf16*)(w + (size_t)64 * 1024 * 1024);        // 32 MB (transposed)
  __bf16* SC = (__bf16*)(w + (size_t)96 * 1024 * 1024);        // 128 MB
  float* stats = (float*)(w + (size_t)224 * 1024 * 1024);      // 128 KB
  float* out = (float*)d_out;

  dim3 blk(256);
  qkv_kernel<<<dim3((Bb * Ss) / BM, Dd / BN, 3), blk, 0, stream>>>(
      x, Wq, Wk, Wv, Qh, Kh, Vt);
  scores_kernel<<<dim3(Ss / BM, Ss / BN, Bb), blk, 0, stream>>>(Qh, Kh, SC);
  stats_kernel<<<dim3(Bb * Ss), blk, 0, stream>>>(SC, stats);
  pv_kernel<<<dim3(Ss / BM, Dd / BN, Bb), blk, 0, stream>>>(SC, stats, Vt, out);
}
